// QBNNAttention_20907900797060
// MI455X (gfx1250) — matrix operand order repaired and checked by reading, not compiled, not run on hardware
//
#include <hip/hip_runtime.h>
#include <hip/hip_bf16.h>
#include <stdint.h>

// ---------------------------------------------------------------------------
// QBNN attention for MI455X (gfx1250, wave32, WMMA).
// All GEMMs use v_wmma_f32_16x16x32_bf16 (16x16 tiles, K=32, f32 accum).
// Attention stages K tiles into LDS with async global->LDS copies
// (double-buffered, ASYNCcnt-tracked) when the toolchain exposes them.
// ---------------------------------------------------------------------------

typedef __bf16 bf16_t;
typedef __attribute__((ext_vector_type(16))) __bf16 v16bf;
typedef __attribute__((ext_vector_type(8)))  __bf16 v8bf;
typedef __attribute__((ext_vector_type(8)))  float  v8f;

#define DEV __device__ __forceinline__

static constexpr int Bz = 2, Sq = 2048, Dm = 1024, Hn = 16, Hd = 64;
static constexpr int BS = Bz * Sq;          // 4096 rows
static constexpr int BH = Bz * Hn;          // 32 head-batches
static constexpr int KLD = 136;             // LDS K-tile row stride (bank-safe)

#if __has_builtin(__builtin_amdgcn_global_load_async_to_lds_b128) && \
    __has_builtin(__builtin_amdgcn_s_wait_asynccnt)
#define HAVE_ASYNC_LDS 1
#else
#define HAVE_ASYNC_LDS 0
#endif

// Builtin parameter types (from the compiler diagnostic):
//   param0: int __attribute__((vector_size(16))) __device__ *   (AS1)
//   param1: same pointee in __shared__ (AS3)
typedef int v4i_t __attribute__((__vector_size__(4 * sizeof(int))));
typedef __attribute__((address_space(1))) v4i_t* gv4i_p;   // global (AS1)
typedef __attribute__((address_space(3))) v4i_t* lv4i_p;   // LDS (AS3)

#define AS1P(p) ((gv4i_p)(uintptr_t)(p))
#define AS3P(p) ((lv4i_p)(uint32_t)(uintptr_t)(p))

DEV v8f wmma_bf16(v16bf a, v16bf b, v8f c) {
  // 8 args: (neg_a, A, neg_b, B, c_mod, C, reuse_a, reuse_b)
  return __builtin_amdgcn_wmma_f32_16x16x32_bf16(
      false, a, false, b, (short)0, c, false, false);
}

DEV v8f zero8() {
  v8f z;
#pragma unroll
  for (int i = 0; i < 8; ++i) z[i] = 0.0f;
  return z;
}

// A-matrix fragment (16x32 bf16, MxK), row-major source, ld in elements.
// Layout: lanes 0-15 = rows, lane half h holds K h*16..h*16+15 contiguous.
DEV v16bf load_a_rm(const bf16_t* __restrict__ p, int ld) {
  int lane = threadIdx.x & 31;
  int r = lane & 15, h = lane >> 4;
  const bf16_t* q = p + (size_t)r * ld + h * 16;
  v8bf lo = *(const v8bf*)(q);
  v8bf hi = *(const v8bf*)(q + 8);
  v16bf f;
#pragma unroll
  for (int i = 0; i < 8; ++i) { f[i] = lo[i]; f[i + 8] = hi[i]; }
  return f;
}

// B-matrix fragment (32x16 bf16, KxN) from *transposed* storage
// Bt (N x K row-major): lane n holds Bt[n][k0 + half*16 .. +15] contiguous.
DEV v16bf load_b_rm(const bf16_t* __restrict__ p, int ld) {
  int lane = threadIdx.x & 31;
  int r = lane & 15, h = lane >> 4;
  return *(const v16bf*)(p + (size_t)r * ld + h * 16);
}

// Same as load_b_rm but via two 16B chunks (for LDS tiles whose row stride
// is only 16B-aligned; lowers to ds_load_b128 pairs).
DEV v16bf load_b_16b(const bf16_t* __restrict__ p, int ld) {
  int lane = threadIdx.x & 31;
  int r = lane & 15, h = lane >> 4;
  const bf16_t* q = p + (size_t)r * ld + h * 16;
  v8bf lo = *(const v8bf*)(q);
  v8bf hi = *(const v8bf*)(q + 8);
  v16bf f;
#pragma unroll
  for (int i = 0; i < 8; ++i) { f[i] = lo[i]; f[i + 8] = hi[i]; }
  return f;
}

// ds_swizzle xor-shuffle within the 32-lane wave (group-of-32 mode).
#define SWZ_XOR(v, m) \
  __uint_as_float(__builtin_amdgcn_ds_swizzle(__float_as_uint(v), (0x1f | ((m) << 10))))

DEV float half16_max(float v) {   // reduce across the 16 lanes of a half-wave
  v = fmaxf(v, SWZ_XOR(v, 1));
  v = fmaxf(v, SWZ_XOR(v, 2));
  v = fmaxf(v, SWZ_XOR(v, 4));
  v = fmaxf(v, SWZ_XOR(v, 8));
  return v;
}
DEV float half16_sum(float v) {
  v += SWZ_XOR(v, 1);
  v += SWZ_XOR(v, 2);
  v += SWZ_XOR(v, 4);
  v += SWZ_XOR(v, 8);
  return v;
}

// ---------------------------------------------------------------------------
// elementwise casts
// ---------------------------------------------------------------------------
__global__ void cast_f32_bf16(const float* __restrict__ src,
                              bf16_t* __restrict__ dst, int n) {
  int i = blockIdx.x * blockDim.x + threadIdx.x;
  if (i < n) dst[i] = (bf16_t)src[i];
}

// Jt[h][e][d] = lambda * J[h][d][e]  (transpose so it loads as a B-fragment)
__global__ void make_jt(const float* __restrict__ J,
                        const float* __restrict__ lambda,
                        bf16_t* __restrict__ Jt) {
  int i = blockIdx.x * blockDim.x + threadIdx.x;
  if (i < Hn * Hd * Hd) {
    int h = i >> 12, e = (i >> 6) & 63, d = i & 63;
    Jt[i] = (bf16_t)(lambda[0] * J[(h << 12) + (d << 6) + e]);
  }
}

// ---------------------------------------------------------------------------
// QKV projection: y[m,e] = sum_d xb[m,d] * W[e,d] + bias[e]
// block = 256 thr (8 waves), block tile 128x128, wave tile 64x32 (4x2 frags).
// MODE 0: Q -> Qcat[..,0:64] = y/8 ; tQ = tanh(y)
// MODE 1: K -> Kcat[..,0:64] = y ; Kcat[..,64:128] = tanh(y)
// MODE 2: V -> Vt[bh][d][s] = y   (transposed for the PV GEMM B-operand)
// ---------------------------------------------------------------------------
template <int MODE>
__global__ __launch_bounds__(256) void proj_kernel(
    const bf16_t* __restrict__ xb, const bf16_t* __restrict__ Wb,
    const float* __restrict__ bias, bf16_t* __restrict__ out_lo,
    bf16_t* __restrict__ out_hi) {
  int tn = blockIdx.x;            // 0..7
  int tm = blockIdx.y;            // 0..31
  int w = threadIdx.x >> 5;
  int wm = w >> 2, wn = w & 3;
  int m0 = tm * 128 + wm * 64;
  int n0 = tn * 128 + wn * 32;

  v8f acc[4][2];
#pragma unroll
  for (int i = 0; i < 4; ++i)
#pragma unroll
    for (int j = 0; j < 2; ++j) acc[i][j] = zero8();

  for (int kk = 0; kk < Dm; kk += 32) {
    v16bf a[4], b[2];
#pragma unroll
    for (int i = 0; i < 4; ++i)
      a[i] = load_a_rm(xb + (size_t)(m0 + 16 * i) * Dm + kk, Dm);
#pragma unroll
    for (int j = 0; j < 2; ++j)
      b[j] = load_b_rm(Wb + (size_t)(n0 + 16 * j) * Dm + kk, Dm);
#pragma unroll
    for (int i = 0; i < 4; ++i)
#pragma unroll
      for (int j = 0; j < 2; ++j) acc[i][j] = wmma_bf16(a[i], b[j], acc[i][j]);
  }

  int lane = threadIdx.x & 31;
  int ncol = lane & 15;
  int mrow = (lane >> 4) * 8;
#pragma unroll
  for (int i = 0; i < 4; ++i) {
#pragma unroll
    for (int j = 0; j < 2; ++j) {
      int e = n0 + 16 * j + ncol;
      float bv = bias[e];
      int h = e >> 6, d = e & 63;
#pragma unroll
      for (int r = 0; r < 8; ++r) {
        int m = m0 + 16 * i + mrow + r;
        int bb = m >> 11;          // / 2048
        int s = m & (Sq - 1);
        int bh = bb * Hn + h;
        float v = acc[i][j][r] + bv;
        if (MODE == 0) {
          out_lo[((size_t)bh * Sq + s) * 128 + d] = (bf16_t)(v * 0.125f);
          out_hi[((size_t)bh * Sq + s) * 64 + d] = (bf16_t)tanhf(v);
        } else if (MODE == 1) {
          out_lo[((size_t)bh * Sq + s) * 128 + d] = (bf16_t)v;
          out_lo[((size_t)bh * Sq + s) * 128 + 64 + d] = (bf16_t)tanhf(v);
        } else {
          out_lo[((size_t)bh * Hd + d) * Sq + s] = (bf16_t)v;  // Vt
        }
      }
    }
  }
}

// ---------------------------------------------------------------------------
// QJ: Qcat[bh][s][64+e] = sum_d tQ[bh][s][d] * Jt[h][e][d]   (lambda folded)
// ---------------------------------------------------------------------------
__global__ __launch_bounds__(256) void qj_kernel(
    const bf16_t* __restrict__ tQ, const bf16_t* __restrict__ Jt,
    bf16_t* __restrict__ Qc) {
  int bh = blockIdx.y;
  int h = bh & (Hn - 1);
  int tm = blockIdx.x;            // 0..7
  int w = threadIdx.x >> 5;
  int wm = w >> 1, wn = w & 1;
  int m0 = tm * 256 + wm * 64;
  int n0 = wn * 32;
  const bf16_t* A = tQ + (size_t)bh * Sq * Hd;
  const bf16_t* Bt = Jt + (size_t)h * Hd * Hd;

  v8f acc[4][2];
#pragma unroll
  for (int i = 0; i < 4; ++i)
#pragma unroll
    for (int j = 0; j < 2; ++j) acc[i][j] = zero8();

#pragma unroll
  for (int kk = 0; kk < Hd; kk += 32) {
    v16bf a[4], b[2];
#pragma unroll
    for (int i = 0; i < 4; ++i)
      a[i] = load_a_rm(A + (size_t)(m0 + 16 * i) * Hd + kk, Hd);
#pragma unroll
    for (int j = 0; j < 2; ++j)
      b[j] = load_b_rm(Bt + (size_t)(n0 + 16 * j) * Hd + kk, Hd);
#pragma unroll
    for (int i = 0; i < 4; ++i)
#pragma unroll
      for (int j = 0; j < 2; ++j) acc[i][j] = wmma_bf16(a[i], b[j], acc[i][j]);
  }

  int lane = threadIdx.x & 31;
  int ncol = lane & 15;
  int mrow = (lane >> 4) * 8;
#pragma unroll
  for (int i = 0; i < 4; ++i)
#pragma unroll
    for (int j = 0; j < 2; ++j)
#pragma unroll
      for (int r = 0; r < 8; ++r) {
        int s = m0 + 16 * i + mrow + r;
        int e = n0 + 16 * j + ncol;
        Qc[((size_t)bh * Sq + s) * 128 + 64 + e] = (bf16_t)acc[i][j][r];
      }
}

// ---------------------------------------------------------------------------
// Flash attention, causal. One block = one (bh, 128-row Q block).
// K tiles (64x128 bf16) are staged into LDS, double-buffered, copied with
// async global->LDS (ASYNCcnt) so the DMA of tile kt+1 overlaps WMMA on kt.
// ---------------------------------------------------------------------------
DEV void copy_k_tile(bf16_t* __restrict__ dst /*LDS*/,
                     const bf16_t* __restrict__ src /*Kcat + j0*128*/) {
  int t = threadIdx.x;
#pragma unroll
  for (int i = 0; i < 4; ++i) {            // 1024 16B chunks / 256 threads
    int c = t + 256 * i;
    int row = c >> 4, colc = c & 15;
    const bf16_t* g = src + (size_t)row * 128 + colc * 8;
    bf16_t* l = dst + (size_t)row * KLD + colc * 8;
#if HAVE_ASYNC_LDS
    __builtin_amdgcn_global_load_async_to_lds_b128(AS1P(g), AS3P(l), 0, 0);
#else
    *(v8bf*)l = *(const v8bf*)g;
#endif
  }
}

__global__ __launch_bounds__(256) void attn_kernel(
    const bf16_t* __restrict__ Qcat, const bf16_t* __restrict__ Kcat,
    const bf16_t* __restrict__ Vt, bf16_t* __restrict__ Ob) {
  __shared__ __align__(16) bf16_t Kl[2][64][KLD];  // 34 KB, double buffer
  __shared__ __align__(32) bf16_t Pl[8][16][80];   // 20 KB, per-wave P tile

  int bh = blockIdx.y;
  int bb = bh >> 4;
  int h = bh & (Hn - 1);
  int qb = blockIdx.x;            // 0..15
  int w = threadIdx.x >> 5;
  int lane = threadIdx.x & 31;
  int ncol = lane & 15;
  int mrow = (lane >> 4) * 8;

  const bf16_t* Qb = Qcat + ((size_t)bh * Sq + qb * 128) * 128;
  const bf16_t* Kb = Kcat + (size_t)bh * Sq * 128;
  const bf16_t* Vb = Vt + (size_t)bh * Hd * Sq;

  // Q fragments for this wave's 16 rows (K=128 -> 4 fragments), loaded once.
  v16bf qf[4];
#pragma unroll
  for (int c = 0; c < 4; ++c)
    qf[c] = load_a_rm(Qb + (size_t)(w * 16) * 128 + c * 32, 128);

  v8f o[4];
#pragma unroll
  for (int f = 0; f < 4; ++f) o[f] = zero8();
  float m_i[8], l_i[8];
#pragma unroll
  for (int r = 0; r < 8; ++r) { m_i[r] = -1.0e30f; l_i[r] = 0.0f; }

  int wave_row0 = qb * 128 + w * 16;
  int ktiles = qb * 2 + 2;        // cover cols 0 .. qb*128+127

  copy_k_tile(&Kl[0][0][0], Kb);  // prologue: tile 0 -> buffer 0

  for (int kt = 0; kt < ktiles; ++kt) {
    int j0 = kt * 64;
#if HAVE_ASYNC_LDS
    __builtin_amdgcn_s_wait_asynccnt(0);
#endif
    __syncthreads();              // K tile kt resident in Kl[kt&1]

    // overlap: start DMA of tile kt+1 into the other buffer
    if (kt + 1 < ktiles) {
      copy_k_tile(&Kl[(kt + 1) & 1][0][0], Kb + (size_t)(j0 + 64) * 128);
      __builtin_prefetch(Vb + j0 + 64, 0, 1);
    }
    const bf16_t* Kt = &Kl[kt & 1][0][0];

    // --- scores: S = Qcat(16x128) @ Kcat_tile^T ---
    v8f sc[4];
#pragma unroll
    for (int f = 0; f < 4; ++f) sc[f] = zero8();
#pragma unroll
    for (int f = 0; f < 4; ++f)
#pragma unroll
      for (int c = 0; c < 4; ++c) {
        v16bf kf = load_b_16b(Kt + (size_t)(f * 16) * KLD + c * 32, KLD);
        sc[f] = wmma_bf16(qf[c], kf, sc[f]);
      }

    // --- causal mask ---
    if (j0 + 63 > wave_row0) {
#pragma unroll
      for (int f = 0; f < 4; ++f)
#pragma unroll
        for (int r = 0; r < 8; ++r) {
          int jg = j0 + f * 16 + ncol;
          int ig = wave_row0 + mrow + r;
          if (jg > ig) sc[f][r] = -1.0e30f;
        }
    }

    // --- online softmax ---
    float mnew[8], corr[8], rsum[8];
#pragma unroll
    for (int r = 0; r < 8; ++r) {
      float mx = sc[0][r];
#pragma unroll
      for (int f = 1; f < 4; ++f) mx = fmaxf(mx, sc[f][r]);
      mx = half16_max(mx);
      mnew[r] = fmaxf(m_i[r], mx);
      corr[r] = __expf(m_i[r] - mnew[r]);
      rsum[r] = 0.0f;
    }
#pragma unroll
    for (int f = 0; f < 4; ++f)
#pragma unroll
      for (int r = 0; r < 8; ++r) {
        float p = __expf(sc[f][r] - mnew[r]);
        sc[f][r] = p;
        rsum[r] += p;
      }
#pragma unroll
    for (int r = 0; r < 8; ++r) {
      l_i[r] = l_i[r] * corr[r] + half16_sum(rsum[r]);
      m_i[r] = mnew[r];
    }
#pragma unroll
    for (int f = 0; f < 4; ++f)
#pragma unroll
      for (int r = 0; r < 8; ++r) o[f][r] *= corr[r];

    // --- stage P through LDS (C-layout -> A-fragment layout), wave-local ---
#pragma unroll
    for (int f = 0; f < 4; ++f)
#pragma unroll
      for (int r = 0; r < 8; ++r)
        Pl[w][mrow + r][f * 16 + ncol] = (bf16_t)sc[f][r];

    v16bf pa[2];
#pragma unroll
    for (int c = 0; c < 2; ++c)
      pa[c] = load_a_rm(&Pl[w][0][0] + c * 32, 80);

    // --- O += P(16x64) @ V ---
#pragma unroll
    for (int f = 0; f < 4; ++f)
#pragma unroll
      for (int c = 0; c < 2; ++c) {
        v16bf vf = load_b_rm(Vb + (size_t)(f * 16) * Sq + j0 + c * 32, Sq);
        o[f] = wmma_bf16(pa[c], vf, o[f]);
      }
  }

  // --- normalize + store (back to [B,S,D] bf16 for the output projection) ---
#pragma unroll
  for (int f = 0; f < 4; ++f)
#pragma unroll
    for (int r = 0; r < 8; ++r) {
      int srow = wave_row0 + mrow + r;
      int d = f * 16 + ncol;
      Ob[((size_t)(bb * Sq + srow)) * Dm + h * Hd + d] =
          (bf16_t)(o[f][r] / l_i[r]);
    }
}

// ---------------------------------------------------------------------------
// Output projection: out[m,e] = sum_d Ob[m,d] * Wo[e,d] + bo[e]   (f32 out)
// ---------------------------------------------------------------------------
__global__ __launch_bounds__(256) void oproj_kernel(
    const bf16_t* __restrict__ Ob, const bf16_t* __restrict__ Wob,
    const float* __restrict__ bo, float* __restrict__ out) {
  int tn = blockIdx.x;
  int tm = blockIdx.y;
  int w = threadIdx.x >> 5;
  int wm = w >> 2, wn = w & 3;
  int m0 = tm * 128 + wm * 64;
  int n0 = tn * 128 + wn * 32;

  v8f acc[4][2];
#pragma unroll
  for (int i = 0; i < 4; ++i)
#pragma unroll
    for (int j = 0; j < 2; ++j) acc[i][j] = zero8();

  for (int kk = 0; kk < Dm; kk += 32) {
    v16bf a[4], b[2];
#pragma unroll
    for (int i = 0; i < 4; ++i)
      a[i] = load_a_rm(Ob + (size_t)(m0 + 16 * i) * Dm + kk, Dm);
#pragma unroll
    for (int j = 0; j < 2; ++j)
      b[j] = load_b_rm(Wob + (size_t)(n0 + 16 * j) * Dm + kk, Dm);
#pragma unroll
    for (int i = 0; i < 4; ++i)
#pragma unroll
      for (int j = 0; j < 2; ++j) acc[i][j] = wmma_bf16(a[i], b[j], acc[i][j]);
  }

  int lane = threadIdx.x & 31;
  int ncol = lane & 15;
  int mrow = (lane >> 4) * 8;
#pragma unroll
  for (int i = 0; i < 4; ++i)
#pragma unroll
    for (int j = 0; j < 2; ++j) {
      int e = n0 + 16 * j + ncol;
      float bv = bo[e];
#pragma unroll
      for (int r = 0; r < 8; ++r) {
        int m = m0 + 16 * i + mrow + r;
        out[(size_t)m * Dm + e] = acc[i][j][r] + bv;
      }
    }
}

// ---------------------------------------------------------------------------
// launch
// ---------------------------------------------------------------------------
extern "C" void kernel_launch(void* const* d_in, const int* in_sizes, int n_in,
                              void* d_out, int out_size, void* d_ws,
                              size_t ws_size, hipStream_t stream) {
  (void)in_sizes; (void)n_in; (void)out_size; (void)ws_size;
  const float* x  = (const float*)d_in[0];
  const float* Wq = (const float*)d_in[1];
  const float* bq = (const float*)d_in[2];
  const float* Wk = (const float*)d_in[3];
  const float* bk = (const float*)d_in[4];
  const float* Wv = (const float*)d_in[5];
  const float* bv = (const float*)d_in[6];
  const float* Wo = (const float*)d_in[7];
  const float* bo = (const float*)d_in[8];
  const float* J  = (const float*)d_in[9];
  const float* lam = (const float*)d_in[10];
  float* out = (float*)d_out;

  char* ws = (char*)d_ws;
  size_t off = 0;
  auto take = [&](size_t bytes) {
    char* p = ws + off;
    off += (bytes + 255) & ~(size_t)255;
    return p;
  };
  bf16_t* xb  = (bf16_t*)take((size_t)BS * Dm * 2);        // 8 MB
  bf16_t* Wqb = (bf16_t*)take((size_t)Dm * Dm * 2);
  bf16_t* Wkb = (bf16_t*)take((size_t)Dm * Dm * 2);
  bf16_t* Wvb = (bf16_t*)take((size_t)Dm * Dm * 2);
  bf16_t* Wob = (bf16_t*)take((size_t)Dm * Dm * 2);
  bf16_t* Jt  = (bf16_t*)take((size_t)Hn * Hd * Hd * 2);
  bf16_t* Qc  = (bf16_t*)take((size_t)BH * Sq * 128 * 2);  // 16 MB
  bf16_t* Kc  = (bf16_t*)take((size_t)BH * Sq * 128 * 2);  // 16 MB
  bf16_t* tQ  = (bf16_t*)take((size_t)BH * Sq * Hd * 2);   // 8 MB
  bf16_t* Vt  = (bf16_t*)take((size_t)BH * Hd * Sq * 2);   // 8 MB
  bf16_t* Ob  = (bf16_t*)take((size_t)BS * Dm * 2);        // 8 MB

  dim3 blk(256);
  cast_f32_bf16<<<(BS * Dm) / 256, blk, 0, stream>>>(x, xb, BS * Dm);
  cast_f32_bf16<<<(Dm * Dm) / 256, blk, 0, stream>>>(Wq, Wqb, Dm * Dm);
  cast_f32_bf16<<<(Dm * Dm) / 256, blk, 0, stream>>>(Wk, Wkb, Dm * Dm);
  cast_f32_bf16<<<(Dm * Dm) / 256, blk, 0, stream>>>(Wv, Wvb, Dm * Dm);
  cast_f32_bf16<<<(Dm * Dm) / 256, blk, 0, stream>>>(Wo, Wob, Dm * Dm);
  make_jt<<<(Hn * Hd * Hd) / 256, blk, 0, stream>>>(J, lam, Jt);

  proj_kernel<0><<<dim3(8, 32), blk, 0, stream>>>(xb, Wqb, bq, Qc, tQ);
  proj_kernel<1><<<dim3(8, 32), blk, 0, stream>>>(xb, Wkb, bk, Kc, nullptr);
  proj_kernel<2><<<dim3(8, 32), blk, 0, stream>>>(xb, Wvb, bv, Vt, nullptr);
  qj_kernel<<<dim3(8, 32), blk, 0, stream>>>(tQ, Jt, Qc);
  attn_kernel<<<dim3(16, 32), blk, 0, stream>>>(Qc, Kc, Vt, Ob);
  oproj_kernel<<<dim3(8, 32), blk, 0, stream>>>(Ob, Wob, bo, out);
}
